// Decoder_55911884259832
// MI455X (gfx1250) — compile-verified, matrix operand order checked by reading, not run on hardware
//
#include <hip/hip_runtime.h>
#include <stdint.h>

// ---------------- problem constants ----------------
constexpr int VSZ   = 32000;   // vocab
constexpr int DIM   = 512;     // E == D == 512
constexpr int BATCH = 16;
constexpr int NBEAM = 4;
constexpr int BB    = BATCH * NBEAM;  // 64 beam rows
constexpr int TSTEP = 16;
constexpr int EOS_TOK = 2;
constexpr int PAD_TOK = 0;
constexpr int KT16 = DIM / 32;        // 16 K-tiles of 32
constexpr long KTSTEP  = 32L * 16;          // fragment elements per kt within an n-tile
constexpr long NTSTRIDE = (long)KT16 * 32 * 16;  // fragment elements per n-tile
#define NEGF (-1e32f)

typedef __attribute__((ext_vector_type(16))) __bf16 v16bf;
typedef __attribute__((ext_vector_type(8)))  float  v8f;

// float -> bf16, round-to-nearest-even
__device__ __forceinline__ __bf16 f2bf(float f) {
  union { float f; uint32_t u; } x; x.f = f;
  uint32_t r = x.u + 0x7FFFu + ((x.u >> 16) & 1u);
  union { uint16_t u; __bf16 b; } y; y.u = (uint16_t)(r >> 16);
  return y.b;
}

// K-index of element e within a 16-bit A/B WMMA fragment (ISA 7.12.2):
// lanes 0-15: VGPR0..3 -> K 0..7, VGPR4..7 -> K 16..23 ; lanes 16-31: +8
__device__ __forceinline__ int k_of_e(int lane, int e) {
  int kbase = (lane < 16) ? 0 : 8;
  int vg = e >> 1;
  int kl = (vg < 4) ? (2 * vg) : (16 + 2 * (vg - 4));
  return kbase + kl + (e & 1);
}

// ------------- f32 -> bf16 fragment-layout conversion (weights AND activations) ------------
// dst layout: tile (nt,kt): ((nt*KT + kt)*32 + lane)*16 + e   (16 bf16 per lane)
// source element (k,n) at W[k*strideK + n*strideN]
__global__ void conv_frag_kernel(const float* __restrict__ W, __bf16* __restrict__ dst,
                                 int KT, long strideK, long strideN, long total) {
  long id = (long)blockIdx.x * blockDim.x + threadIdx.x;
  if (id >= total) return;
  int  e    = (int)(id & 15);
  int  lane = (int)((id >> 4) & 31);
  long tk   = id >> 9;
  int  kt   = (int)(tk % KT);
  long nt   = tk / KT;
  long n = nt * 16 + (lane & 15);
  long k = (long)kt * 32 + k_of_e(lane, e);
  dst[id] = f2bf(W[k * strideK + n * strideN]);
}

// embedding gather straight into A-fragment layout (skips the f32 X buffer)
__global__ void embed_frag_kernel(const float* __restrict__ emb, const int* __restrict__ TOK,
                                  __bf16* __restrict__ XF) {
  int id = blockIdx.x * blockDim.x + threadIdx.x;
  if (id >= BB * DIM) return;            // 4 rb * 16 kt * 32 lane * 16 e = 32768
  int e    = id & 15;
  int lane = (id >> 4) & 31;
  int kt   = (id >> 9) & 15;
  int rb   = id >> 13;
  int m = rb * 16 + (lane & 15);
  int k = kt * 32 + k_of_e(lane, e);
  XF[id] = f2bf(emb[(long)TOK[m] * DIM + k]);
}

// ------------- wave GEMM: 16 rows x (NT*16) cols, software-pipelined fragments + WMMA ----
template<int NT>
__device__ __forceinline__ void gemm_frag(v8f* acc, const __bf16* __restrict__ AF,
                                          const __bf16* __restrict__ BF,
                                          int rb, int ct0, int lane) {
  const __bf16* aptr  = AF + (((long)rb * KT16) * 32 + lane) * 16;
  const __bf16* bbase = BF + (((long)ct0 * KT16) * 32 + lane) * 16;
  v16bf a = *(const v16bf*)aptr;
  for (int kt = 0; kt < KT16; ++kt) {
    const __bf16* bk = bbase + (long)kt * KTSTEP;
    v16bf bcur = *(const v16bf*)bk;
    v16bf anext = a;
    if (kt + 1 < KT16) {
      anext = *(const v16bf*)(aptr + (long)(kt + 1) * KTSTEP);
      __builtin_prefetch((const void*)(bk + KTSTEP), 0, 3);  // next K-tile of this B panel
    }
#pragma unroll
    for (int n = 0; n < NT; ++n) {
      v16bf bnext = bcur;
      if (n + 1 < NT) bnext = *(const v16bf*)(bk + (long)(n + 1) * NTSTRIDE);
      acc[n] = __builtin_amdgcn_wmma_f32_16x16x32_bf16(false, a, false, bcur,
                                                       (short)0, acc[n], false, false);
      bcur = bnext;
    }
    a = anext;
  }
}

// z[64x2048] = x @ WihT + h @ WhhT + b  (two fragment GEMMs accumulated)
__global__ void lstm_gemm_kernel(const __bf16* __restrict__ XF, const __bf16* __restrict__ WiF,
                                 const __bf16* __restrict__ HF, const __bf16* __restrict__ WhF,
                                 const float* __restrict__ bias, float* __restrict__ z) {
  int lane = threadIdx.x & 31;
  int wid  = (int)((blockIdx.x * blockDim.x + threadIdx.x) >> 5);
  int rb   = wid & 3;        // 4 row-blocks of 16
  int g    = wid >> 2;       // 0..31 -> 4 tiles each over N=2048
  int ct0  = g * 4;
  v8f acc[4] = {};
  gemm_frag<4>(acc, XF, WiF, rb, ct0, lane);
  gemm_frag<4>(acc, HF, WhF, rb, ct0, lane);
#pragma unroll
  for (int n = 0; n < 4; ++n)
#pragma unroll
    for (int i = 0; i < 8; ++i) {
      int row = rb * 16 + i + ((lane >= 16) ? 8 : 0);
      int col = (ct0 + n) * 16 + (lane & 15);
      z[(long)row * (4 * DIM) + col] = acc[n][i] + bias[col];
    }
}

// logits[64x32000] = h1 @ Wp + bp
__global__ void proj_kernel(const __bf16* __restrict__ HF, const __bf16* __restrict__ WPF,
                            const float* __restrict__ bp, float* __restrict__ out) {
  int lane = threadIdx.x & 31;
  int wid  = (int)((blockIdx.x * blockDim.x + threadIdx.x) >> 5);
  int rb   = wid & 3;
  int g    = wid >> 2;       // 0..499 -> 4 tiles each over N=32000
  int ct0  = g * 4;
  v8f acc[4] = {};
  gemm_frag<4>(acc, HF, WPF, rb, ct0, lane);
#pragma unroll
  for (int n = 0; n < 4; ++n)
#pragma unroll
    for (int i = 0; i < 8; ++i) {
      int row = rb * 16 + i + ((lane >= 16) ? 8 : 0);
      int col = (ct0 + n) * 16 + (lane & 15);
      out[(long)row * VSZ + col] = acc[n][i] + bp[col];
    }
}

// ------------- elementwise / small kernels -------------
__global__ void init_kernel(const float* __restrict__ h0, const float* __restrict__ c0,
                            const int* __restrict__ inputs,
                            float* __restrict__ H, float* __restrict__ C, int* __restrict__ TOK) {
  int id = blockIdx.x * blockDim.x + threadIdx.x;
  if (id < 2 * BB * DIM) {
    int d  = id & (DIM - 1);
    int bb = (id >> 9) & 63;
    int l  = id >> 15;
    long src = ((long)l * BATCH + (bb >> 2)) * DIM + d;
    H[id] = h0[src];
    C[id] = c0[src];
  }
  if (id < BB) TOK[id] = inputs[id >> 2];
}

__global__ void gates_kernel(const float* __restrict__ z, float* __restrict__ h,
                             float* __restrict__ c) {
  int id = blockIdx.x * blockDim.x + threadIdx.x;
  if (id >= BB * DIM) return;
  int d  = id & (DIM - 1);
  int bb = id >> 9;
  const float* zr = z + (long)bb * (4 * DIM);
  float zi = zr[d], zf = zr[d + DIM], zg = zr[d + 2 * DIM], zo = zr[d + 3 * DIM];
  float cp = c[id];
  float si = 1.f / (1.f + expf(-zi));
  float sf = 1.f / (1.f + expf(-zf));
  float so = 1.f / (1.f + expf(-zo));
  float cn = sf * cp + si * tanhf(zg);
  c[id] = cn;
  h[id] = so * tanhf(cn);
}

__global__ void lse_kernel(const float* __restrict__ logits, float* __restrict__ lse) {
  __shared__ float red[256];
  int row = blockIdx.x, tid = threadIdx.x;
  const float* p = logits + (long)row * VSZ;
  float m = -3.4e38f;
  for (int v = tid; v < VSZ; v += 256) m = fmaxf(m, p[v]);
  red[tid] = m; __syncthreads();
  for (int s = 128; s > 0; s >>= 1) { if (tid < s) red[tid] = fmaxf(red[tid], red[tid + s]); __syncthreads(); }
  m = red[0]; __syncthreads();
  float sum = 0.f;
  for (int v = tid; v < VSZ; v += 256) sum += expf(p[v] - m);
  red[tid] = sum; __syncthreads();
  for (int s = 128; s > 0; s >>= 1) { if (tid < s) red[tid] += red[tid + s]; __syncthreads(); }
  if (tid == 0) lse[row] = m + logf(red[0]);
}

__device__ __forceinline__ bool better(float v1, int i1, float v2, int i2) {
  return (v1 > v2) || (v1 == v2 && i1 < i2);
}

__global__ void topk_kernel(const float* __restrict__ logits, const float* __restrict__ lse,
                            float* __restrict__ LP, const int* __restrict__ FINo,
                            int* __restrict__ PIDX, int* __restrict__ TOK, int t0) {
  __shared__ float sv[1024];
  __shared__ int   si[1024];
  int b = blockIdx.x, tid = threadIdx.x;
  float lp[NBEAM] = {0.f, 0.f, 0.f, 0.f};
  int   fin[NBEAM] = {0, 0, 0, 0};
  if (!t0) {
#pragma unroll
    for (int j = 0; j < NBEAM; ++j) { lp[j] = LP[b * NBEAM + j]; fin[j] = FINo[b * NBEAM + j]; }
  }
  float bv[4] = {-3.4e38f, -3.4e38f, -3.4e38f, -3.4e38f};
  int   bi[4] = {0x7fffffff, 0x7fffffff, 0x7fffffff, 0x7fffffff};
  int nj = t0 ? 1 : NBEAM;
  for (int j = 0; j < nj; ++j) {
    int row = b * NBEAM + j;
    const float* lg = logits + (long)row * VSZ;
    float base = t0 ? (-lse[row]) : (lp[j] - lse[row]);
    bool fdone = (!t0) && (fin[j] != 0);
    for (int v = tid; v < VSZ; v += 256) {
      float val = fdone ? ((v == PAD_TOK) ? lp[j] : NEGF) : (lg[v] + base);
      int idx = j * VSZ + v;
      if (better(val, idx, bv[3], bi[3])) {
        bv[3] = val; bi[3] = idx;
#pragma unroll
        for (int k = 3; k > 0; --k)
          if (better(bv[k], bi[k], bv[k - 1], bi[k - 1])) {
            float tv = bv[k]; bv[k] = bv[k - 1]; bv[k - 1] = tv;
            int   ti = bi[k]; bi[k] = bi[k - 1]; bi[k - 1] = ti;
          }
      }
    }
  }
#pragma unroll
  for (int k = 0; k < 4; ++k) { sv[tid * 4 + k] = bv[k]; si[tid * 4 + k] = bi[k]; }
  __syncthreads();
  if (tid == 0) {
    for (int j = 0; j < NBEAM; ++j) {
      float bestv = -3.4e38f; int besti = 0x7fffffff; int bestpos = 0;
      for (int q = 0; q < 1024; ++q)
        if (better(sv[q], si[q], bestv, besti)) { bestv = sv[q]; besti = si[q]; bestpos = q; }
      sv[bestpos] = -3.4e38f; si[bestpos] = 0x7fffffff;
      LP[b * NBEAM + j]   = bestv;
      PIDX[b * NBEAM + j] = besti / VSZ + b * NBEAM;
      TOK[b * NBEAM + j]  = besti % VSZ;
    }
  }
}

__global__ void gather_hc_kernel(const float* __restrict__ Ho, float* __restrict__ Hn,
                                 const float* __restrict__ Co, float* __restrict__ Cn,
                                 const int* __restrict__ PIDX) {
  int id = blockIdx.x * blockDim.x + threadIdx.x;
  if (id >= 2 * BB * DIM) return;
  int d  = id & (DIM - 1);
  int bb = (id >> 9) & 63;
  int l  = id >> 15;
  long src = ((long)l * BB + PIDX[bb]) * DIM + d;
  Hn[id] = Ho[src];
  Cn[id] = Co[src];
}

__global__ void update_beams_kernel(const int* __restrict__ BOo, int* __restrict__ BOn,
                                    const int* __restrict__ FINo, int* __restrict__ FINn,
                                    const int* __restrict__ PIDX, const int* __restrict__ TOK,
                                    const int* __restrict__ inputs, int t, int t0) {
  int bb = threadIdx.x;
  if (bb >= BB) return;
  int tok = TOK[bb], p = PIDX[bb];
  int fo = t0 ? 0 : FINo[p];
  FINn[bb] = fo | (tok == EOS_TOK);
  for (int r = 0; r <= TSTEP; ++r) BOn[r * BB + bb] = t0 ? 0 : BOo[r * BB + p];
  if (t0) BOn[bb] = inputs[bb >> 2];
  BOn[(t + 1) * BB + bb] = tok;
}

__global__ void finalize_kernel(const int* __restrict__ BO, float* __restrict__ out) {
  int id = blockIdx.x * blockDim.x + threadIdx.x;
  if (id >= (TSTEP + 1) * BB) return;
  out[id] = (float)BO[id];
}

// ---------------- host launch ----------------
extern "C" void kernel_launch(void* const* d_in, const int* in_sizes, int n_in,
                              void* d_out, int out_size, void* d_ws, size_t ws_size,
                              hipStream_t stream) {
  (void)in_sizes; (void)n_in; (void)out_size; (void)ws_size;
  const int*   inputs = (const int*)d_in[0];
  const float* h0   = (const float*)d_in[1];
  const float* c0   = (const float*)d_in[2];
  const float* emb  = (const float*)d_in[3];
  const float* Wih0 = (const float*)d_in[4];
  const float* Whh0 = (const float*)d_in[5];
  const float* b0   = (const float*)d_in[6];
  const float* Wih1 = (const float*)d_in[7];
  const float* Whh1 = (const float*)d_in[8];
  const float* b1   = (const float*)d_in[9];
  const float* Wp   = (const float*)d_in[10];
  const float* bp   = (const float*)d_in[11];
  float* out = (float*)d_out;

  size_t off = 0;
  char* wsb = (char*)d_ws;
  auto take = [&](size_t bytes) -> void* {
    void* p = wsb + off;
    off += (bytes + 255) & ~(size_t)255;
    return p;
  };
  __bf16* WPF  = (__bf16*)take((size_t)VSZ * DIM * 2);
  __bf16* WF0i = (__bf16*)take((size_t)4 * DIM * DIM * 2);
  __bf16* WF0h = (__bf16*)take((size_t)4 * DIM * DIM * 2);
  __bf16* WF1i = (__bf16*)take((size_t)4 * DIM * DIM * 2);
  __bf16* WF1h = (__bf16*)take((size_t)4 * DIM * DIM * 2);
  __bf16* XF   = (__bf16*)take((size_t)BB * DIM * 2);   // A-fragments: embedded input
  __bf16* AH0F = (__bf16*)take((size_t)BB * DIM * 2);   // A-fragments: layer-0 h / h0n
  __bf16* AH1F = (__bf16*)take((size_t)BB * DIM * 2);   // A-fragments: layer-1 h / h1n
  float* H   = (float*)take((size_t)2 * 2 * BB * DIM * 4);  // 2 ping-pong slots
  float* C   = (float*)take((size_t)2 * 2 * BB * DIM * 4);
  float* Z   = (float*)take((size_t)BB * 4 * DIM * 4);
  float* LSE = (float*)take(256);
  float* LP  = (float*)take(256);
  int* PIDX  = (int*)take(256);
  int* TOK   = (int*)take(256);
  int* BO    = (int*)take((size_t)2 * (TSTEP + 1) * BB * 4);
  int* FIN   = (int*)take((size_t)2 * BB * 4);

  // one-time (per launch) weight conversions into WMMA fragment layout
  {
    long totWp = (long)VSZ * DIM;
    conv_frag_kernel<<<(int)((totWp + 255) / 256), 256, 0, stream>>>(
        Wp, WPF, KT16, (long)VSZ, 1L, totWp);
    long totW = (long)4 * DIM * DIM;
    int blk = (int)((totW + 255) / 256);
    conv_frag_kernel<<<blk, 256, 0, stream>>>(Wih0, WF0i, KT16, 1L, (long)DIM, totW);
    conv_frag_kernel<<<blk, 256, 0, stream>>>(Whh0, WF0h, KT16, 1L, (long)DIM, totW);
    conv_frag_kernel<<<blk, 256, 0, stream>>>(Wih1, WF1i, KT16, 1L, (long)DIM, totW);
    conv_frag_kernel<<<blk, 256, 0, stream>>>(Whh1, WF1h, KT16, 1L, (long)DIM, totW);
  }

  init_kernel<<<(2 * BB * DIM + 255) / 256, 256, 0, stream>>>(h0, c0, inputs, H, C, TOK);

  const size_t HS = (size_t)2 * BB * DIM;   // floats per ping-pong slot
  const long totA = (long)BB * DIM;         // 32768 fragment elements per activation
  const int  blkA = (int)((totA + 255) / 256);
  for (int t = 0; t < TSTEP; ++t) {
    int s = t & 1, ns = s ^ 1;
    float* Hs = H + (size_t)s * HS;  float* Hn = H + (size_t)ns * HS;
    float* Cs = C + (size_t)s * HS;  float* Cn = C + (size_t)ns * HS;
    float* lg = out + (size_t)t * BB * VSZ;
    int t0 = (t == 0) ? 1 : 0;

    // layer 0: A-fragments for x (fused embed) and h0
    embed_frag_kernel<<<blkA, 256, 0, stream>>>(emb, TOK, XF);
    conv_frag_kernel<<<blkA, 256, 0, stream>>>(Hs, AH0F, KT16, 1L, (long)DIM, totA);
    lstm_gemm_kernel<<<32, 128, 0, stream>>>(XF, WF0i, AH0F, WF0h, b0, Z);
    gates_kernel<<<(BB * DIM + 255) / 256, 256, 0, stream>>>(Z, Hs, Cs);
    // layer 1: A-fragments for h0n and old h1
    conv_frag_kernel<<<blkA, 256, 0, stream>>>(Hs, AH0F, KT16, 1L, (long)DIM, totA);
    conv_frag_kernel<<<blkA, 256, 0, stream>>>(Hs + BB * DIM, AH1F, KT16, 1L, (long)DIM, totA);
    lstm_gemm_kernel<<<32, 128, 0, stream>>>(AH0F, WF1i, AH1F, WF1h, b1, Z);
    gates_kernel<<<(BB * DIM + 255) / 256, 256, 0, stream>>>(Z, Hs + BB * DIM, Cs + BB * DIM);
    // projection from h1n fragments
    conv_frag_kernel<<<blkA, 256, 0, stream>>>(Hs + BB * DIM, AH1F, KT16, 1L, (long)DIM, totA);
    proj_kernel<<<500, 128, 0, stream>>>(AH1F, WPF, bp, lg);

    lse_kernel<<<BB, 256, 0, stream>>>(lg, LSE);
    topk_kernel<<<BATCH, 256, 0, stream>>>(lg, LSE, LP, FIN + s * BB, PIDX, TOK, t0);
    gather_hc_kernel<<<(2 * BB * DIM + 255) / 256, 256, 0, stream>>>(Hs, Hn, Cs, Cn, PIDX);
    update_beams_kernel<<<1, 64, 0, stream>>>(BO + s * (TSTEP + 1) * BB,
                                              BO + ns * (TSTEP + 1) * BB,
                                              FIN + s * BB, FIN + ns * BB,
                                              PIDX, TOK, inputs, t, t0);
  }
  // after t=15, final state lives in slot 0
  finalize_kernel<<<((TSTEP + 1) * BB + 255) / 256, 256, 0, stream>>>(
      BO, out + (size_t)TSTEP * BB * VSZ);
}
